// ConformanceGNN_4002909520798
// MI455X (gfx1250) — compile-verified, matrix operand order checked by reading, not run on hardware
//
#include <hip/hip_runtime.h>
#include <hip/hip_bf16.h>
#include <stdint.h>

// ---------------------------------------------------------------------------
// Problem constants (from reference)
// ---------------------------------------------------------------------------
#define PN 50000
#define TN 20000
#define EN 250000
#define HD 128
#define LN 3
#define TFD 8
#define PFXD 18

typedef __attribute__((ext_vector_type(16))) __bf16 v16bf;
typedef __attribute__((ext_vector_type(8)))  float  v8f;

union Frag {
    uint4    q[2];
    unsigned u[8];
    v16bf    v;
};

__device__ __forceinline__ unsigned short f2bf(float f) {
    unsigned u = __float_as_uint(f);
    unsigned r = u + 0x7FFFu + ((u >> 16) & 1u);   // round-to-nearest-even
    return (unsigned short)(r >> 16);
}
__device__ __forceinline__ unsigned fenc(float f) {  // order-preserving encode
    unsigned u = __float_as_uint(f);
    return (u & 0x80000000u) ? ~u : (u | 0x80000000u);
}
__device__ __forceinline__ float fdec(unsigned e) {
    return (e & 0x80000000u) ? __uint_as_float(e & 0x7FFFFFFFu)
                             : __uint_as_float(~e);
}

// ---------------------------------------------------------------------------
// Small utility kernels
// ---------------------------------------------------------------------------
__global__ void k_zero(float* p, int n) {
    for (int i = blockIdx.x * blockDim.x + threadIdx.x; i < n;
         i += gridDim.x * blockDim.x) p[i] = 0.0f;
}
__global__ void k_copy(float* d, const float* s, int n) {
    int i = blockIdx.x * blockDim.x + threadIdx.x;
    if (i < n) d[i] = s[i];
}
__global__ void k_cvt_bf16(const float* __restrict__ x,
                           unsigned short* __restrict__ y, int n) {
    for (int i = blockIdx.x * blockDim.x + threadIdx.x; i < n;
         i += gridDim.x * blockDim.x) y[i] = f2bf(x[i]);
}
// pack fp32 weight [K,128] -> bf16-pair dwords [K/2][128] (B-fragment friendly)
__global__ void k_packw(const float* __restrict__ W, unsigned* __restrict__ out,
                        int K) {
    int n = (K >> 1) * HD;
    for (int i = blockIdx.x * blockDim.x + threadIdx.x; i < n;
         i += gridDim.x * blockDim.x) {
        int kp = i >> 7, c = i & 127;
        unsigned lo = f2bf(W[(size_t)(2 * kp) * HD + c]);
        unsigned hi = f2bf(W[(size_t)(2 * kp + 1) * HD + c]);
        out[i] = lo | (hi << 16);
    }
}

// ---------------------------------------------------------------------------
// Embeddings
// ---------------------------------------------------------------------------
__global__ void k_embed_place(const float* __restrict__ pf,
                              const float* __restrict__ W,
                              const float* __restrict__ b,
                              float* __restrict__ H,
                              unsigned short* __restrict__ Hb) {
    int i = blockIdx.x * blockDim.x + threadIdx.x;
    if (i >= PN * HD) return;
    int p = i >> 7, h = i & 127;
    float v = pf[p] * W[h] + b[h];
    H[i] = v; Hb[i] = f2bf(v);
}
__global__ void k_embed_trans(const float* __restrict__ tf,
                              const float* __restrict__ W,
                              const float* __restrict__ b,
                              float* __restrict__ H,
                              unsigned short* __restrict__ Hb) {
    int t = blockIdx.x, h = threadIdx.x;
    float s = b[h];
    #pragma unroll
    for (int j = 0; j < TFD; ++j) s += tf[t * TFD + j] * W[j * HD + h];
    H[(size_t)t * HD + h] = s; Hb[(size_t)t * HD + h] = f2bf(s);
}

// ---------------------------------------------------------------------------
// Attention fold:  v = W @ wa  ;  c = b . wa + ba
// ---------------------------------------------------------------------------
__global__ void k_fold(const float* __restrict__ W, const float* __restrict__ b,
                       const float* __restrict__ wa, const float* __restrict__ ba,
                       float* __restrict__ v, float* __restrict__ cOut) {
    int i = threadIdx.x;                       // blockDim = 128
    float s = 0.f;
    for (int j = 0; j < HD; ++j) s += W[(size_t)i * HD + j] * wa[j];
    v[i] = s;
    if (i == 0) {
        float c = ba[0];
        for (int j = 0; j < HD; ++j) c += b[j] * wa[j];
        cOut[0] = c;
    }
}

// logit[e] = H[src[e]] . v + c     (folded attention score)
__global__ void k_logit(const float* __restrict__ H, const int* __restrict__ src,
                        const float* __restrict__ v, const float* __restrict__ c,
                        float* __restrict__ out, int E) {
    int e = blockIdx.x * blockDim.x + threadIdx.x;
    if (e >= E) return;
    const float4* row = (const float4*)(H + (size_t)src[e] * HD);
    const float4* v4  = (const float4*)v;
    float s = c[0];
    #pragma unroll 8
    for (int i = 0; i < HD / 4; ++i) {
        float4 a = row[i], b = v4[i];
        s += a.x * b.x + a.y * b.y + a.z * b.z + a.w * b.w;
    }
    out[e] = s;
}

// ---------------------------------------------------------------------------
// Global softmax reductions over E edges
// ---------------------------------------------------------------------------
__global__ void k_initred(unsigned* slot, float* sum) {
    slot[0] = 0u;      // encoded -inf
    sum[0]  = 0.0f;
}
__global__ void k_redmax(const float* __restrict__ x, int n, unsigned* slot) {
    __shared__ float sm[256];
    float m = -3.4e38f;
    for (int i = blockIdx.x * blockDim.x + threadIdx.x; i < n;
         i += gridDim.x * blockDim.x) m = fmaxf(m, x[i]);
    sm[threadIdx.x] = m; __syncthreads();
    for (int s = blockDim.x >> 1; s; s >>= 1) {
        if (threadIdx.x < s) sm[threadIdx.x] = fmaxf(sm[threadIdx.x], sm[threadIdx.x + s]);
        __syncthreads();
    }
    if (threadIdx.x == 0) atomicMax(slot, fenc(sm[0]));
}
__global__ void k_prob(const float* __restrict__ x, const unsigned* __restrict__ slot,
                       float* __restrict__ p, float* __restrict__ sum, int n) {
    __shared__ float sm[256];
    float m = fdec(slot[0]);
    float loc = 0.f;
    for (int i = blockIdx.x * blockDim.x + threadIdx.x; i < n;
         i += gridDim.x * blockDim.x) {
        float v = __expf(x[i] - m);
        p[i] = v; loc += v;
    }
    sm[threadIdx.x] = loc; __syncthreads();
    for (int s = blockDim.x >> 1; s; s >>= 1) {
        if (threadIdx.x < s) sm[threadIdx.x] += sm[threadIdx.x + s];
        __syncthreads();
    }
    if (threadIdx.x == 0) atomicAdd(sum, sm[0]);
}

// ---------------------------------------------------------------------------
// WMMA GEMM (bf16 inputs, fp32 accumulate), 128-wide output.
//   C[r,:] = gather(A,idx)[r,:] @ W (+bias)      K = KS*32 (KS = 4 or 8)
// Compile-time modes (no runtime branching in the hot loop):
//   GATHER : A rows indirected through gidx
//   SCATTER: Agg[dst[r]] += C[r] * prob[r]/sum  (fused softmax segment-sum)
//            otherwise C is stored to Cout.
// Block = 128 threads = 4 waves; each wave owns 2 of the 8 N-tiles; the block
// grid-strides over 16-row tiles. B fragments hoisted into VGPRs across the
// entire row loop; A fragments loaded as one 8x b128 clause per 4 k-steps.
// The gather index is software-pipelined one iteration ahead and the next A
// row is prefetched (global_prefetch_b8) under the WMMA shadow.
// ---------------------------------------------------------------------------
template <int KS, bool GATHER, bool SCATTER>
__global__ void __launch_bounds__(128)
k_gemm_wmma(const unsigned short* __restrict__ A0,   // bf16 [.,128]
            const unsigned short* __restrict__ A1,   // bf16 [.,128] (K=256 tail)
            const int* __restrict__ gidx,            // gather indices (GATHER)
            const unsigned* __restrict__ Wpk,        // packed bf16 pairs [KS*16][128]
            const float* __restrict__ bias,          // [128]
            float* __restrict__ Cout,                // [R][128]   (!SCATTER)
            const float* __restrict__ prob,          // [R]        (SCATTER)
            const float* __restrict__ psum,          // scalar     (SCATTER)
            const int* __restrict__ dst,             // [R]        (SCATTER)
            float* __restrict__ Agg,                 // [*][128]   (SCATTER)
            int R)
{
    const int lane = threadIdx.x & 31;
    const int wave = threadIdx.x >> 5;
    const int hsel = (lane & 16) ? 1 : 0;   // upper half-wave
    const int lrow = lane & 15;
    const int numTiles = R >> 4;

    const float invS = SCATTER ? (1.0f / psum[0]) : 0.0f;
    // per-lane output columns for this wave's two N-tiles (invariant)
    const int col0 = wave * 32 + lrow;
    const float b0 = bias[col0];
    const float b1 = bias[col0 + 16];

    // Hoist B fragments (weights) for this wave's two N-tiles.
    Frag B[2][KS];
    #pragma unroll
    for (int t = 0; t < 2; ++t) {
        const int col = wave * 32 + t * 16 + lrow;
        #pragma unroll
        for (int kt = 0; kt < KS; ++kt) {
            const int kpb = kt * 16 + (hsel ? 4 : 0);
            #pragma unroll
            for (int i = 0; i < 4; ++i) {
                B[t][kt].u[i]     = Wpk[(kpb + i) * HD + col];
                B[t][kt].u[4 + i] = Wpk[(kpb + 8 + i) * HD + col];
            }
        }
    }

    // software-pipelined gather index (one grid-stride iteration ahead)
    int rt = blockIdx.x;
    int arow = 0;
    if (rt < numTiles) arow = GATHER ? gidx[rt * 16 + lrow] : rt * 16 + lrow;

    for (; rt < numTiles; rt += gridDim.x) {
        const int cur = arow;
        const int nrt = rt + gridDim.x;           // uniform across the block
        if (nrt < numTiles) {
            // issue next tile's index load early; latency hidden by the WMMAs
            arow = GATHER ? gidx[nrt * 16 + lrow] : nrt * 16 + lrow;
        }

        const unsigned short* a0p = A0 + (size_t)cur * HD;
        const unsigned short* a1p = (KS > 4) ? (A1 + (size_t)cur * HD) : a0p;

        v8f acc0, acc1;
        #pragma unroll
        for (int i = 0; i < 8; ++i) { acc0[i] = 0.f; acc1[i] = 0.f; }

        #pragma unroll
        for (int kc = 0; kc < KS; kc += 4) {
            // one clause of 8 x b128 loads covering 4 k-steps
            Frag A[4];
            #pragma unroll
            for (int k = 0; k < 4; ++k) {
                const unsigned short* src = ((kc + k) < 4) ? a0p : a1p;
                const int kk = ((kc + k) & 3) * 32 + (hsel ? 8 : 0);
                A[k].q[0] = *(const uint4*)(src + kk);
                A[k].q[1] = *(const uint4*)(src + kk + 16);
            }
            #pragma unroll
            for (int k = 0; k < 4; ++k) {
                acc0 = __builtin_amdgcn_wmma_f32_16x16x32_bf16(
                    false, A[k].v, false, B[0][kc + k].v, (short)0, acc0, false, false);
                acc1 = __builtin_amdgcn_wmma_f32_16x16x32_bf16(
                    false, A[k].v, false, B[1][kc + k].v, (short)0, acc1, false, false);
            }
        }

        // prefetch next tile's gathered A row (global_prefetch_b8) while the
        // epilogue retires; pulls the random 256B row toward the caches.
        if (nrt < numTiles) {
            __builtin_prefetch(A0 + (size_t)arow * HD, 0, 3);
            if (KS > 4) __builtin_prefetch(A1 + (size_t)arow * HD, 0, 3);
        }

        if (SCATTER) {
            #pragma unroll
            for (int j = 0; j < 8; ++j) {
                const int orow = rt * 16 + j + (hsel ? 8 : 0);
                const float cf = prob[orow] * invS;
                float* base = Agg + (size_t)dst[orow] * HD + col0;
                atomicAdd(base,      (acc0[j] + b0) * cf);
                atomicAdd(base + 16, (acc1[j] + b1) * cf);
            }
        } else {
            #pragma unroll
            for (int j = 0; j < 8; ++j) {
                const int orow = rt * 16 + j + (hsel ? 8 : 0);
                float* base = Cout + (size_t)orow * HD + col0;
                base[0]  = acc0[j] + b0;
                base[16] = acc1[j] + b1;
            }
        }
    }
}

// residual + relu, refresh bf16 mirror
__global__ void k_resrelu(float* __restrict__ h, const float* __restrict__ nb,
                          unsigned short* __restrict__ hb, int n) {
    for (int i = blockIdx.x * blockDim.x + threadIdx.x; i < n;
         i += gridDim.x * blockDim.x) {
        float v = h[i] + nb[i];
        v = v > 0.f ? v : 0.f;
        h[i] = v; hb[i] = f2bf(v);
    }
}

// column mean (scaled column sum via one atomic per block per column)
__global__ void k_colsum(const float* __restrict__ X, float* __restrict__ out,
                         float scale, int R) {
    int c = threadIdx.x;                 // blockDim = 128
    int r0 = blockIdx.x * 64;
    int r1 = r0 + 64; if (r1 > R) r1 = R;
    float s = 0.f;
    for (int r = r0; r < r1; ++r) s += X[(size_t)r * HD + c];
    atomicAdd(out + c, s * scale);
}

// generic small dense layer: y[n] = act( x . W[:,n] + b[n] ),  W row-major [K,N]
__global__ void k_dense(const float* __restrict__ x, const float* __restrict__ W,
                        const float* __restrict__ b, float* __restrict__ y,
                        float* __restrict__ y2, int K, int N, int act) {
    for (int n = blockIdx.x * blockDim.x + threadIdx.x; n < N;
         n += gridDim.x * blockDim.x) {
        float s = b ? b[n] : 0.f;
        for (int k = 0; k < K; ++k) s += x[k] * W[(size_t)k * N + n];
        if (act == 1)      s = fmaxf(s, 0.f);
        else if (act == 2) s = 1.f / (1.f + __expf(-s));
        y[n] = s;
        if (y2) y2[n] = s;
    }
}

// enabled mask: viol[t] += (marking[src[e]] <= 0) ; enabled = viol==0
__global__ void k_enab_scatter(const float* __restrict__ mk,
                               const int* __restrict__ src,
                               const int* __restrict__ dst,
                               float* __restrict__ viol, int E) {
    int e = blockIdx.x * blockDim.x + threadIdx.x;
    if (e >= E) return;
    if (mk[src[e]] <= 0.f) atomicAdd(viol + dst[e], 1.f);
}
__global__ void k_enab(const float* __restrict__ viol, float* __restrict__ y,
                       float* __restrict__ y2, int T) {
    int t = blockIdx.x * blockDim.x + threadIdx.x;
    if (t >= T) return;
    float v = (viol[t] == 0.f) ? 1.f : 0.f;
    y[t] = v; y2[t] = v;
}

// ---------------------------------------------------------------------------
// Host orchestration
// ---------------------------------------------------------------------------
static inline int cdiv(long long a, long long b) { return (int)((a + b - 1) / b); }

extern "C" void kernel_launch(void* const* d_in, const int* in_sizes, int n_in,
                              void* d_out, int out_size, void* d_ws, size_t ws_size,
                              hipStream_t stream) {
    (void)in_sizes; (void)n_in; (void)out_size; (void)ws_size;
    // ---- inputs (setup_inputs dict order) ----
    const float* place_features = (const float*)d_in[0];
    const float* trans_features = (const float*)d_in[1];
    const float* prefix_enc     = (const float*)d_in[2];
    const int*   pre_ei         = (const int*)d_in[3];   // [2,E]: row0 place, row1 transition
    const int*   post_ei        = (const int*)d_in[4];   // [2,E]: row0 transition, row1 place
    const float* W_pe = (const float*)d_in[5];  const float* b_pe = (const float*)d_in[6];
    const float* W_te = (const float*)d_in[7];  const float* b_te = (const float*)d_in[8];
    const float* W_px = (const float*)d_in[9];  const float* b_px = (const float*)d_in[10];
    const float* Wp2t = (const float*)d_in[11]; const float* bp2t = (const float*)d_in[12];
    const float* Wt2p = (const float*)d_in[13]; const float* bt2p = (const float*)d_in[14];
    const float* Wpu  = (const float*)d_in[15]; const float* bpu  = (const float*)d_in[16];
    const float* Wtu  = (const float*)d_in[17]; const float* btu  = (const float*)d_in[18];
    const float* Wpa  = (const float*)d_in[19]; const float* bpa  = (const float*)d_in[20];
    const float* Wta  = (const float*)d_in[21]; const float* bta  = (const float*)d_in[22];
    const float* Wpp  = (const float*)d_in[23]; const float* bpp  = (const float*)d_in[24];
    const float* Wtp  = (const float*)d_in[25]; const float* btp  = (const float*)d_in[26];
    const float* W1   = (const float*)d_in[27]; const float* b1   = (const float*)d_in[28];
    const float* W2   = (const float*)d_in[29]; const float* b2   = (const float*)d_in[30];
    const float* W3   = (const float*)d_in[31]; const float* b3   = (const float*)d_in[32];
    const float* Wc1  = (const float*)d_in[33]; const float* bc1  = (const float*)d_in[34];
    const float* Wc2  = (const float*)d_in[35]; const float* bc2  = (const float*)d_in[36];
    const float* Wc3  = (const float*)d_in[37]; const float* bc3  = (const float*)d_in[38];

    // ---- workspace bump allocator ----
    char* wsp = (char*)d_ws;
    size_t off = 0;
    auto alloc = [&](size_t bytes) -> void* {
        void* p = wsp + off;
        off = (off + bytes + 255) & ~(size_t)255;
        return p;
    };
    float*          place_h  = (float*)alloc((size_t)PN * HD * 4);
    float*          trans_h  = (float*)alloc((size_t)TN * HD * 4);
    unsigned short* place_hb = (unsigned short*)alloc((size_t)PN * HD * 2);
    unsigned short* trans_hb = (unsigned short*)alloc((size_t)TN * HD * 2);
    float*          aggP     = (float*)alloc((size_t)PN * HD * 4);
    unsigned short* aggPb    = (unsigned short*)alloc((size_t)PN * HD * 2);
    float*          aggT     = (float*)alloc((size_t)TN * HD * 4);
    unsigned short* aggTb    = (unsigned short*)alloc((size_t)TN * HD * 2);
    float*          newbuf   = (float*)alloc((size_t)PN * HD * 4);
    float*          logits   = (float*)alloc((size_t)EN * 4);
    float*          probs    = (float*)alloc((size_t)EN * 4);
    float*          red      = (float*)alloc(64 * 4);      // [0]=enc max, [1]=sum, [2]=c
    float*          vfold    = (float*)alloc(HD * 4);
    float*          combined = (float*)alloc(384 * 4);
    float*          cin      = (float*)alloc((384 + 2 * TN) * 4);
    float*          h1       = (float*)alloc(256 * 4);
    float*          h2       = (float*)alloc(128 * 4);
    float*          c1       = (float*)alloc(256 * 4);
    float*          c2       = (float*)alloc(128 * 4);
    float*          viol     = (float*)alloc((size_t)TN * 4);
    unsigned*       pkP2T    = (unsigned*)alloc((size_t)LN * 64 * HD * 4);
    unsigned*       pkT2P    = (unsigned*)alloc((size_t)LN * 64 * HD * 4);
    unsigned*       pkPU     = (unsigned*)alloc((size_t)LN * 128 * HD * 4);
    unsigned*       pkTU     = (unsigned*)alloc((size_t)LN * 128 * HD * 4);
    unsigned*       pkPP     = (unsigned*)alloc((size_t)64 * HD * 4);
    unsigned*       pkTP     = (unsigned*)alloc((size_t)64 * HD * 4);
    float*          d_outf   = (float*)d_out;

    const int GB = 8192;   // grid cap for GEMM row-tile striding

    // ---- pack weights to bf16-pair layout ----
    for (int l = 0; l < LN; ++l) {
        k_packw<<<cdiv(64 * HD, 256), 256, 0, stream>>>(Wp2t + (size_t)l * HD * HD, pkP2T + (size_t)l * 64 * HD, HD);
        k_packw<<<cdiv(64 * HD, 256), 256, 0, stream>>>(Wt2p + (size_t)l * HD * HD, pkT2P + (size_t)l * 64 * HD, HD);
        k_packw<<<cdiv(128 * HD, 256), 256, 0, stream>>>(Wpu + (size_t)l * 2 * HD * HD, pkPU + (size_t)l * 128 * HD, 2 * HD);
        k_packw<<<cdiv(128 * HD, 256), 256, 0, stream>>>(Wtu + (size_t)l * 2 * HD * HD, pkTU + (size_t)l * 128 * HD, 2 * HD);
    }
    k_packw<<<cdiv(64 * HD, 256), 256, 0, stream>>>(Wpp, pkPP, HD);
    k_packw<<<cdiv(64 * HD, 256), 256, 0, stream>>>(Wtp, pkTP, HD);

    // ---- embeddings ----
    k_embed_place<<<cdiv((size_t)PN * HD, 256), 256, 0, stream>>>(place_features, W_pe, b_pe, place_h, place_hb);
    k_embed_trans<<<TN, HD, 0, stream>>>(trans_features, W_te, b_te, trans_h, trans_hb);

    // ---- GNN layers ----
    for (int l = 0; l < LN; ++l) {
        // place -> transition messages (folded-attention softmax + fused scatter GEMM)
        k_fold<<<1, HD, 0, stream>>>(Wp2t + (size_t)l * HD * HD, bp2t + l * HD,
                                     Wta + l * HD, bta + l, vfold, red + 2);
        k_logit<<<cdiv(EN, 256), 256, 0, stream>>>(place_h, pre_ei, vfold, red + 2, logits, EN);
        k_initred<<<1, 1, 0, stream>>>((unsigned*)red, red + 1);
        k_redmax<<<1024, 256, 0, stream>>>(logits, EN, (unsigned*)red);
        k_prob<<<1024, 256, 0, stream>>>(logits, (unsigned*)red, probs, red + 1, EN);
        k_zero<<<cdiv((size_t)TN * HD, 256), 256, 0, stream>>>(aggT, TN * HD);
        k_gemm_wmma<4, true, true><<<GB, 128, 0, stream>>>(
            place_hb, nullptr, pre_ei, pkP2T + (size_t)l * 64 * HD, bp2t + l * HD,
            nullptr, probs, red + 1, pre_ei + EN, aggT, EN);

        // transition -> place messages
        k_fold<<<1, HD, 0, stream>>>(Wt2p + (size_t)l * HD * HD, bt2p + l * HD,
                                     Wpa + l * HD, bpa + l, vfold, red + 2);
        k_logit<<<cdiv(EN, 256), 256, 0, stream>>>(trans_h, post_ei, vfold, red + 2, logits, EN);
        k_initred<<<1, 1, 0, stream>>>((unsigned*)red, red + 1);
        k_redmax<<<1024, 256, 0, stream>>>(logits, EN, (unsigned*)red);
        k_prob<<<1024, 256, 0, stream>>>(logits, (unsigned*)red, probs, red + 1, EN);
        k_zero<<<cdiv((size_t)PN * HD, 256), 256, 0, stream>>>(aggP, PN * HD);
        k_gemm_wmma<4, true, true><<<GB, 128, 0, stream>>>(
            trans_hb, nullptr, post_ei, pkT2P + (size_t)l * 64 * HD, bt2p + l * HD,
            nullptr, probs, red + 1, post_ei + EN, aggP, EN);

        // bf16 mirrors of aggregated messages
        k_cvt_bf16<<<cdiv((size_t)PN * HD, 256), 256, 0, stream>>>(aggP, aggPb, PN * HD);
        k_cvt_bf16<<<cdiv((size_t)TN * HD, 256), 256, 0, stream>>>(aggT, aggTb, TN * HD);

        // node updates: concat[h, msgs] @ W (K=256) + residual ReLU
        k_gemm_wmma<8, false, false><<<GB, 128, 0, stream>>>(
            place_hb, aggPb, nullptr, pkPU + (size_t)l * 128 * HD, bpu + l * HD,
            newbuf, nullptr, nullptr, nullptr, nullptr, PN);
        k_resrelu<<<cdiv((size_t)PN * HD, 256), 256, 0, stream>>>(place_h, newbuf, place_hb, PN * HD);
        k_gemm_wmma<8, false, false><<<GB, 128, 0, stream>>>(
            trans_hb, aggTb, nullptr, pkTU + (size_t)l * 128 * HD, btu + l * HD,
            newbuf, nullptr, nullptr, nullptr, nullptr, TN);
        k_resrelu<<<cdiv((size_t)TN * HD, 256), 256, 0, stream>>>(trans_h, newbuf, trans_hb, TN * HD);
    }

    // ---- global pooling ----
    k_zero<<<2, 256, 0, stream>>>(combined, 384);
    k_gemm_wmma<4, false, false><<<GB, 128, 0, stream>>>(
        place_hb, nullptr, nullptr, pkPP, bpp,
        newbuf, nullptr, nullptr, nullptr, nullptr, PN);
    k_colsum<<<cdiv(PN, 64), 128, 0, stream>>>(newbuf, combined, 1.0f / PN, PN);
    k_gemm_wmma<4, false, false><<<GB, 128, 0, stream>>>(
        trans_hb, nullptr, nullptr, pkTP, btp,
        newbuf, nullptr, nullptr, nullptr, nullptr, TN);
    k_colsum<<<cdiv(TN, 64), 128, 0, stream>>>(newbuf, combined + 128, 1.0f / TN, TN);
    k_dense<<<1, 128, 0, stream>>>(prefix_enc, W_px, b_px, combined + 256, nullptr, PFXD, HD, 0);

    // ---- prediction head ----
    k_copy<<<2, 256, 0, stream>>>(cin, combined, 384);
    k_dense<<<1, 256, 0, stream>>>(combined, W1, b1, h1, nullptr, 384, 256, 1);
    k_dense<<<1, 128, 0, stream>>>(h1, W2, b2, h2, nullptr, 256, 128, 1);
    // next_transitions -> d_out[0:T) and cin[384:384+T)
    k_dense<<<cdiv(TN, 256), 256, 0, stream>>>(h2, W3, b3, cin + 384, d_outf, 128, TN, 2);

    // ---- enabled mask -> d_out[T+1 : T+1+T) and cin tail ----
    k_zero<<<cdiv(TN, 256), 256, 0, stream>>>(viol, TN);
    k_enab_scatter<<<cdiv(EN, 256), 256, 0, stream>>>(place_features, pre_ei, pre_ei + EN, viol, EN);
    k_enab<<<cdiv(TN, 256), 256, 0, stream>>>(viol, cin + 384 + TN, d_outf + TN + 1, TN);

    // ---- conformance head -> d_out[T] ----
    k_dense<<<1, 256, 0, stream>>>(cin, Wc1, bc1, c1, nullptr, 384 + 2 * TN, 256, 1);
    k_dense<<<1, 128, 0, stream>>>(c1, Wc2, bc2, c2, nullptr, 256, 128, 1);
    k_dense<<<1, 1, 0, stream>>>(c2, Wc3, bc3, d_outf + TN, nullptr, 128, 1, 2);
}
